// SelfAttention_12343736008831
// MI455X (gfx1250) — matrix operand order repaired and checked by reading, not compiled, not run on hardware
//
#include <hip/hip_runtime.h>
#include <hip/hip_bf16.h>

typedef __bf16 bf16;
typedef __attribute__((ext_vector_type(16))) __bf16 v16bf;
typedef __attribute__((ext_vector_type(8)))  __bf16 v8bf;
typedef __attribute__((ext_vector_type(8)))  float  v8f;
typedef __attribute__((ext_vector_type(4)))  int    v4i;

#define B_  8
#define T_  2048
#define E_  1024
#define H_  64
#define TA_ 2112   // key-dim stride padding (kept for layout stability)

// ---- workspace layout (bytes) ----
#define WS_WTQ 0u
#define WS_WTK 131072u            // 64*1024*2
#define WS_WTV 262144u
#define WS_QB  393216u            // Q  bf16 [B*T][H]
#define WS_KB  2490368u           // K  bf16 [B][TA][H]
#define WS_VT  4653056u           // Vt bf16 [B][H][TA]

#if defined(__has_builtin)
#if __has_builtin(__builtin_amdgcn_global_load_async_to_lds_b128)
#define USE_ASYNC_LDS 1
#endif
#endif

__device__ __forceinline__ v8f wmma_bf16(v16bf a, v16bf b, v8f c) {
    return __builtin_amdgcn_wmma_f32_16x16x32_bf16(false, a, false, b,
                                                   (short)0, c, false, false);
}
__device__ __forceinline__ v16bf cat8(v8bf lo, v8bf hi) {
    return __builtin_shufflevector(lo, hi, 0,1,2,3,4,5,6,7,8,9,10,11,12,13,14,15);
}

// 16B global -> LDS copy. Async (ASYNCcnt) on gfx1250 when the builtin exists.
// Builtin takes AS1/AS3 pointers to 16B (v4i) payloads; flat->AS1 is
// representation-identical and flat LDS addr[31:0] is the LDS offset (ISA 10.2).
__device__ __forceinline__ void cp16_g2l(const bf16* g, bf16* l) {
#ifdef USE_ASYNC_LDS
    __builtin_amdgcn_global_load_async_to_lds_b128(
        (__attribute__((address_space(1))) v4i*)(uintptr_t)g,
        (__attribute__((address_space(3))) v4i*)(uintptr_t)l, 0, 0);
#else
    *(v8bf*)l = *(const v8bf*)g;                   // fallback: register round-trip
#endif
}
__device__ __forceinline__ void staging_wait() {
#ifdef USE_ASYNC_LDS
#if __has_builtin(__builtin_amdgcn_s_wait_asynccnt)
    __builtin_amdgcn_s_wait_asynccnt(0);
#else
    asm volatile("s_wait_asynccnt 0x0" ::: "memory");
#endif
#endif
}

// ---------------- kernel 1: W (E x H, f32) -> Wt (H x E, bf16) ----------------
__global__ void wt_kernel(const float* __restrict__ Wq, const float* __restrict__ Wk,
                          const float* __restrict__ Wv,
                          bf16* __restrict__ wtq, bf16* __restrict__ wtk,
                          bf16* __restrict__ wtv) {
    int i = blockIdx.x * blockDim.x + threadIdx.x;     // i = h*E + e
    if (i >= H_ * E_) return;
    int h = i >> 10, e = i & (E_ - 1);
    wtq[i] = (bf16)Wq[e * H_ + h];
    wtk[i] = (bf16)Wk[e * H_ + h];
    wtv[i] = (bf16)Wv[e * H_ + h];
}

// ---------------- kernel 2: fused QKV projection (bf16 WMMA, f32 accum) -------
__global__ __launch_bounds__(256)
void qkv_kernel(const float* __restrict__ x,
                const bf16* __restrict__ wtq, const bf16* __restrict__ wtk,
                const bf16* __restrict__ wtv,
                bf16* __restrict__ qb, bf16* __restrict__ kb, bf16* __restrict__ vt) {
    const int wave = threadIdx.x >> 5, lane = threadIdx.x & 31;
    const int half = lane >> 4, ln = lane & 15;
    const int row0 = blockIdx.x * 128 + wave * 16;     // flat row over B*T

    v8f accq[4], acck[4], accv[4];
    #pragma unroll
    for (int c = 0; c < 4; ++c) { accq[c] = (v8f){}; acck[c] = (v8f){}; accv[c] = (v8f){}; }

    const float* xr = x + (size_t)(row0 + ln) * E_;
    for (int es = 0; es < E_ / 32; ++es) {
        v8f f0 = *(const v8f*)(xr + es * 32 + half * 16);
        v8f f1 = *(const v8f*)(xr + es * 32 + half * 16 + 8);
        v16bf a = cat8(__builtin_convertvector(f0, v8bf),
                       __builtin_convertvector(f1, v8bf));
        #pragma unroll
        for (int c = 0; c < 4; ++c) {
            const int wofs = (c * 16 + ln) * E_ + es * 32 + half * 16;
            v16bf bq = *(const v16bf*)(wtq + wofs);
            v16bf bk = *(const v16bf*)(wtk + wofs);
            v16bf bv = *(const v16bf*)(wtv + wofs);
            accq[c] = wmma_bf16(a, bq, accq[c]);
            acck[c] = wmma_bf16(a, bk, acck[c]);
            accv[c] = wmma_bf16(a, bv, accv[c]);
        }
    }

    const int b = row0 >> 11, t0 = row0 & (T_ - 1);
    #pragma unroll
    for (int c = 0; c < 4; ++c) {
        const int h = c * 16 + ln;
        #pragma unroll
        for (int r = 0; r < 8; ++r) {
            const int row = row0 + r + 8 * half;
            const int t   = t0   + r + 8 * half;
            qb[(size_t)row * H_ + h]           = (bf16)(accq[c][r] * 0.125f); // fold 1/sqrt(H)
            kb[((size_t)b * TA_ + t) * H_ + h] = (bf16)acck[c][r];
        }
        v8bf vv = __builtin_convertvector(accv[c], v8bf);
        *(v8bf*)(vt + ((size_t)(b * H_ + h)) * TA_ + t0 + 8 * half) = vv;
    }
}

// ---------------- kernel 3: causal flash attention --------------------------
// 64-key tiles staged once per block into double-buffered LDS via async
// global->LDS copies; all 8 waves consume from LDS. Uniform trip count across
// waves (block max), per-wave compute skip keeps barriers deadlock-free.
__global__ __launch_bounds__(256)
void attn_kernel(const bf16* __restrict__ qb, const bf16* __restrict__ kb,
                 const bf16* __restrict__ vt, float* __restrict__ out) {
    __shared__ __align__(32) bf16 Ks[2][64][64];       // [buf][key][h]   8KB each
    __shared__ __align__(32) bf16 Vs[2][64][64];       // [buf][h][key]   8KB each
    __shared__ __align__(16) bf16 plds[8][16][64];     // per-wave P tile 2KB each

    const int wave = threadIdx.x >> 5, lane = threadIdx.x & 31;
    const int half = lane >> 4, ln = lane & 15;
    const int b = blockIdx.x >> 4;
    const int qtile = blockIdx.x & 15;
    const int q0 = qtile * 128 + wave * 16;

    const bf16* kbase = kb + (size_t)b * TA_ * H_;
    const bf16* vbase = vt + (size_t)b * H_ * TA_;

    // staging work split: 256 threads x 4 x 16B = 16KB per tile (K + V)
    const int skey = threadIdx.x >> 3;                  // 0..31
    const int sj   = (threadIdx.x & 7) * 8;             // element offset in 64-wide row
    auto stage = [&](int bs, int kt2) {
        const int k0 = kt2 * 64;
        const bf16* gk = kbase + (size_t)(k0 + skey) * H_ + sj;
        cp16_g2l(gk,           &Ks[bs][skey][sj]);
        cp16_g2l(gk + 32 * H_, &Ks[bs][skey + 32][sj]);
        const bf16* gv = vbase + (size_t)skey * TA_ + k0 + sj;
        cp16_g2l(gv,                    &Vs[bs][skey][sj]);
        cp16_g2l(gv + (size_t)32 * TA_, &Vs[bs][skey + 32][sj]);
    };

    // Q A-fragments, resident in VGPRs for the whole loop
    const bf16* qrow = qb + (size_t)(b * T_ + q0 + ln) * H_;
    v16bf qa[2];
    #pragma unroll
    for (int hs = 0; hs < 2; ++hs)
        qa[hs] = cat8(*(const v8bf*)(qrow + hs * 32 + half * 16),
                      *(const v8bf*)(qrow + hs * 32 + half * 16 + 8));

    v8f o[4];
    #pragma unroll
    for (int c = 0; c < 4; ++c) o[c] = (v8f){};
    float mrun[8], lrun[8];
    #pragma unroll
    for (int r = 0; r < 8; ++r) { mrun[r] = -1e30f; lrun[r] = 0.f; }

    const int ntw = (q0 + 79) >> 6;                    // tiles this wave needs
    const int ntb = (qtile * 128 + 191) >> 6;          // block max (wave 7)

    stage(0, 0);
    for (int kt = 0; kt < ntb; ++kt) {
        const int bufi = kt & 1;
        staging_wait();                                // own async copies landed
        __syncthreads();                               // all waves' copies landed
        if (kt + 1 < ntb) stage(bufi ^ 1, kt + 1);     // prior reads of buf^1 done

        if (kt < ntw) {                                // wave-uniform: EXEC stays full
            const int k0 = kt * 64;
            // S = (Q/8) K^T : 4 key chunks x 2 h-steps
            v8f s[4];
            #pragma unroll
            for (int c = 0; c < 4; ++c) {
                s[c] = (v8f){};
                const bf16* kr = &Ks[bufi][c * 16 + ln][0];
                #pragma unroll
                for (int hs = 0; hs < 2; ++hs)
                    s[c] = wmma_bf16(qa[hs], *(const v16bf*)(kr + hs * 32 + half * 16), s[c]);
            }
            if (k0 + 63 > q0) {                        // causal mask on boundary tiles
                #pragma unroll
                for (int c = 0; c < 4; ++c) {
                    const int key = k0 + c * 16 + ln;
                    #pragma unroll
                    for (int r = 0; r < 8; ++r) {
                        const int row = q0 + r + 8 * half;
                        s[c][r] = (key > row) ? -1e30f : s[c][r];
                    }
                }
            }
            // online softmax, per-row stats across 16 lanes of a half-group
            #pragma unroll
            for (int r = 0; r < 8; ++r) {
                float mx = fmaxf(fmaxf(s[0][r], s[1][r]), fmaxf(s[2][r], s[3][r]));
                mx = fmaxf(mx, __shfl_xor(mx, 1));
                mx = fmaxf(mx, __shfl_xor(mx, 2));
                mx = fmaxf(mx, __shfl_xor(mx, 4));
                mx = fmaxf(mx, __shfl_xor(mx, 8));
                const float mn = fmaxf(mrun[r], mx);
                const float alpha = __expf(mrun[r] - mn);
                mrun[r] = mn;
                float p[4], rs = 0.f;
                #pragma unroll
                for (int c = 0; c < 4; ++c) { p[c] = __expf(s[c][r] - mn); rs += p[c]; }
                rs += __shfl_xor(rs, 1);
                rs += __shfl_xor(rs, 2);
                rs += __shfl_xor(rs, 4);
                rs += __shfl_xor(rs, 8);
                lrun[r] = lrun[r] * alpha + rs;
                o[0][r] *= alpha; o[1][r] *= alpha; o[2][r] *= alpha; o[3][r] *= alpha;
                #pragma unroll
                for (int c = 0; c < 4; ++c)
                    plds[wave][r + 8 * half][c * 16 + ln] = (bf16)p[c];
            }
            asm volatile("s_wait_dscnt 0" ::: "memory");   // same-wave DS RAW fence

            // P A-frags (16x64 keys), then O += P * V from staged V^T
            v16bf pf0 = cat8(*(const v8bf*)&plds[wave][ln][half * 16],
                             *(const v8bf*)&plds[wave][ln][half * 16 + 8]);
            v16bf pf1 = cat8(*(const v8bf*)&plds[wave][ln][32 + half * 16],
                             *(const v8bf*)&plds[wave][ln][32 + half * 16 + 8]);
            #pragma unroll
            for (int c = 0; c < 4; ++c) {
                const bf16* vr = &Vs[bufi][c * 16 + ln][0];
                o[c] = wmma_bf16(pf0, *(const v16bf*)(vr + half * 16), o[c]);
                o[c] = wmma_bf16(pf1, *(const v16bf*)(vr + 32 + half * 16), o[c]);
            }
        }
    }

    #pragma unroll
    for (int c = 0; c < 4; ++c) {
        const int h = c * 16 + ln;
        #pragma unroll
        for (int r = 0; r < 8; ++r) {
            const int row = q0 + r + 8 * half;
            out[(size_t)(b * T_ + row) * H_ + h] = o[c][r] / lrun[r];
        }
    }
}

extern "C" void kernel_launch(void* const* d_in, const int* in_sizes, int n_in,
                              void* d_out, int out_size, void* d_ws, size_t ws_size,
                              hipStream_t stream) {
    (void)in_sizes; (void)n_in; (void)out_size; (void)ws_size;
    const float* x  = (const float*)d_in[0];
    const float* Wq = (const float*)d_in[1];
    const float* Wk = (const float*)d_in[2];
    const float* Wv = (const float*)d_in[3];
    float* out = (float*)d_out;
    char* ws = (char*)d_ws;

    bf16* wtq = (bf16*)(ws + WS_WTQ);
    bf16* wtk = (bf16*)(ws + WS_WTK);
    bf16* wtv = (bf16*)(ws + WS_WTV);
    bf16* qb  = (bf16*)(ws + WS_QB);
    bf16* kb  = (bf16*)(ws + WS_KB);
    bf16* vt  = (bf16*)(ws + WS_VT);

    wt_kernel<<<(H_ * E_ + 255) / 256, 256, 0, stream>>>(Wq, Wk, Wv, wtq, wtk, wtv);
    qkv_kernel<<<(B_ * T_) / 128, 256, 0, stream>>>(x, wtq, wtk, wtv, qb, kb, vt);
    attn_kernel<<<B_ * (T_ / 128), 256, 0, stream>>>(qb, kb, vt, out);
}
